// SpatialSelfAttention_1906965479813
// MI455X (gfx1250) — matrix operand order repaired and checked by reading, not compiled, not run on hardware
//
#include <hip/hip_runtime.h>
#include <math.h>
#include <stdint.h>

// ---------------- problem constants ----------------
constexpr int BN   = 2;      // batch
constexpr int CC   = 512;    // channels
constexpr int HW   = 4096;   // 64*64 tokens
constexpr int GRP  = 32;     // groups
constexpr int CPG  = CC / GRP; // 16 channels / group
constexpr float EPS = 1e-6f;

typedef _Float16 half_t;
typedef __attribute__((ext_vector_type(16))) _Float16 v16h;
typedef __attribute__((ext_vector_type(8)))  _Float16 v8h;
typedef __attribute__((ext_vector_type(8)))  float    v8f;

// CDNA5 async global->LDS path (ASYNCcnt), guarded: falls back to
// synchronous global_load + ds_store if the builtins are absent.
#if __has_builtin(__builtin_amdgcn_global_load_async_to_lds_b128) && \
    __has_builtin(__builtin_amdgcn_s_wait_asynccnt)
#define USE_ASYNC_LDS 1
#else
#define USE_ASYNC_LDS 0
#endif

// Builtin parameter types (from hipcc diagnostic):
//   arg0: int __vector_size__(16) * in device (global, AS1) space
//   arg1: same pointee in shared (LDS, AS3) space
typedef __attribute__((__vector_size__(4 * sizeof(int)))) int v4i_t;
typedef __attribute__((address_space(1))) v4i_t* gv4i_p;
typedef __attribute__((address_space(3))) v4i_t* lv4i_p;

// =====================================================================
// Kernel 1: per-(batch, group) mean / rstd.  Channels of a group are
// contiguous in NCHW -> each block reduces one contiguous 256KB span.
// =====================================================================
__global__ __launch_bounds__(256) void gn_stats(const float* __restrict__ x,
                                                float* __restrict__ meanrstd) {
  const int bg = blockIdx.x;                 // 0 .. BN*GRP-1
  const float* base = x + (size_t)bg * CPG * HW;
  const int count = CPG * HW;                // 65536
  float s = 0.f, ss = 0.f;
  for (int i = threadIdx.x; i < count; i += 256) {
    float v = base[i];
    s += v; ss += v * v;
  }
  __shared__ float rs[256], rss[256];
  rs[threadIdx.x] = s; rss[threadIdx.x] = ss;
  __syncthreads();
  for (int st = 128; st > 0; st >>= 1) {
    if (threadIdx.x < st) {
      rs[threadIdx.x]  += rs[threadIdx.x + st];
      rss[threadIdx.x] += rss[threadIdx.x + st];
    }
    __syncthreads();
  }
  if (threadIdx.x == 0) {
    float m   = rs[0] / (float)count;
    float var = rss[0] / (float)count - m * m;
    meanrstd[2 * bg]     = m;
    meanrstd[2 * bg + 1] = rsqrtf(var + EPS);
  }
}

// =====================================================================
// Kernel 2: normalize + affine, write f16 activations in (b, n, c)
// GEMM layout (c fastest -> coalesced stores).
// =====================================================================
__global__ __launch_bounds__(256) void gn_apply(const float* __restrict__ x,
                                                const float* __restrict__ gamma,
                                                const float* __restrict__ beta,
                                                const float* __restrict__ meanrstd,
                                                half_t* __restrict__ t) {
  size_t idx = (size_t)blockIdx.x * 256 + threadIdx.x;
  if (idx >= (size_t)BN * HW * CC) return;
  int c = (int)(idx & (CC - 1));
  size_t rest = idx >> 9;
  int n = (int)(rest & (HW - 1));
  int b = (int)(rest >> 12);
  int g = c >> 4;
  float m = meanrstd[2 * (b * GRP + g)];
  float r = meanrstd[2 * (b * GRP + g) + 1];
  float v = x[((size_t)b * CC + c) * HW + n];
  t[idx] = (half_t)((v - m) * r * gamma[c] + beta[c]);
}

// =====================================================================
// Kernel 3: f32 -> f16 weight conversion
// =====================================================================
__global__ __launch_bounds__(256) void f32_to_f16(const float* __restrict__ s,
                                                  half_t* __restrict__ d, int n) {
  int i = blockIdx.x * 256 + threadIdx.x;
  if (i < n) d[i] = (half_t)s[i];
}

// =====================================================================
// Generic WMMA GEMM:   out = alpha * (A @ Bsrc^T) + bias [+ residual]
//   A    : (M, K) f16 row-major, ld = lda
//   Bsrc : (Ncols, K) f16 row-major (so B operand = Bsrc^T, contiguous-K reads)
// Block tile 128x128, 8 waves (4Mx2N) of 32x64 each -> 8 WMMAs / K-step.
// Double-buffered LDS; stage via async global->LDS (ASYNCcnt) when available.
// MODE 0: f16 out row-major (ld = ldo)
// MODE 1: f16 out transposed per batch: out[b][col][n]  (V -> V^T)
// MODE 2: f32 out NCHW + residual:      out[b][col][n] = res + val
// =====================================================================
constexpr int TS = 40;  // LDS tile row stride in halfs (80B rows -> conflict-free frag reads)

template <int MODE>
__global__ __launch_bounds__(256) void gemm_wmma(
    const half_t* __restrict__ A, int lda,
    const half_t* __restrict__ Bsrc, int ldb,
    const float* __restrict__ bias, float alpha,
    void* __restrict__ outp, int ldo, int Ksz,
    const float* __restrict__ residual) {
  __shared__ half_t At[2][128 * TS];
  __shared__ half_t Bt[2][128 * TS];

  const int tid   = threadIdx.x;
  const int lane  = tid & 31;
  const int wave  = tid >> 5;
  const int waveM = (wave >> 1) * 32;   // 4 M-waves
  const int waveN = (wave & 1) * 64;    // 2 N-waves
  const int blockM = blockIdx.x * 128;
  const int blockN = blockIdx.y * 128;

  // cooperative-load assignment: each thread owns 16 halfs (32B) of each tile
  const int lrow = tid >> 1;
  const int lseg = (tid & 1) * 16;
  const half_t* gA = A    + (size_t)(blockM + lrow) * lda + lseg;
  const half_t* gB = Bsrc + (size_t)(blockN + lrow) * ldb + lseg;
  const int lofs = lrow * TS + lseg;

  // stage tile (k0) into LDS buffer `buf`
  auto stage = [&](int buf, int k0) {
#if USE_ASYNC_LDS
    __builtin_amdgcn_global_load_async_to_lds_b128(
        (gv4i_p)(uintptr_t)(gA + k0),
        (lv4i_p)(uintptr_t)&At[buf][lofs], 0, 0);
    __builtin_amdgcn_global_load_async_to_lds_b128(
        (gv4i_p)(uintptr_t)(gA + k0 + 8),
        (lv4i_p)(uintptr_t)&At[buf][lofs + 8], 0, 0);
    __builtin_amdgcn_global_load_async_to_lds_b128(
        (gv4i_p)(uintptr_t)(gB + k0),
        (lv4i_p)(uintptr_t)&Bt[buf][lofs], 0, 0);
    __builtin_amdgcn_global_load_async_to_lds_b128(
        (gv4i_p)(uintptr_t)(gB + k0 + 8),
        (lv4i_p)(uintptr_t)&Bt[buf][lofs + 8], 0, 0);
#else
    v8h a0 = *(const v8h*)(gA + k0);
    v8h a1 = *(const v8h*)(gA + k0 + 8);
    v8h b0 = *(const v8h*)(gB + k0);
    v8h b1 = *(const v8h*)(gB + k0 + 8);
    *(v8h*)&At[buf][lofs]     = a0;
    *(v8h*)&At[buf][lofs + 8] = a1;
    *(v8h*)&Bt[buf][lofs]     = b0;
    *(v8h*)&Bt[buf][lofs + 8] = b1;
    if (k0 + 32 < Ksz) {
      __builtin_prefetch(gA + k0 + 32, 0, 1);   // global_prefetch_b8
      __builtin_prefetch(gB + k0 + 32, 0, 1);
    }
#endif
  };

  // ISA fragment layout constants (wave32)
  const int m16 = lane & 15;            // A row within 16
  const int kbA = (lane >> 4) * 8;      // A: lanes 0-15 K=0-7/16-23, lanes 16-31 K=8-15/24-31
  const int n16 = lane & 15;            // B col within 16
  const int kbB = (lane >> 4) * 8;      // B: lanes 0-15 K=0-7/16-23, lanes 16-31 K=8-15/24-31

  v8f acc[2][4];
#pragma unroll
  for (int mi = 0; mi < 2; ++mi)
#pragma unroll
    for (int ni = 0; ni < 4; ++ni)
      acc[mi][ni] = (v8f){0.f, 0.f, 0.f, 0.f, 0.f, 0.f, 0.f, 0.f};

  stage(0, 0);
  int cur = 0;
  for (int k0 = 0; k0 < Ksz; k0 += 32) {
    if (k0 + 32 < Ksz) {
      stage(cur ^ 1, k0 + 32);     // overlap next stage with current compute
#if USE_ASYNC_LDS
      __builtin_amdgcn_s_wait_asynccnt(4);  // previous stage (oldest 4 ops) done
#endif
    } else {
#if USE_ASYNC_LDS
      __builtin_amdgcn_s_wait_asynccnt(0);
#endif
    }
    __syncthreads();  // current buffer visible to all waves

    // ---- fragments from LDS (ds_load_b128) ----
    v16h afr[2], bfr[4];
#pragma unroll
    for (int mi = 0; mi < 2; ++mi) {
      int rb = (waveM + mi * 16 + m16) * TS;
      v8h lo = *(const v8h*)&At[cur][rb + kbA];
      v8h hi = *(const v8h*)&At[cur][rb + 16 + kbA];
#pragma unroll
      for (int i = 0; i < 8; ++i) { afr[mi][i] = lo[i]; afr[mi][8 + i] = hi[i]; }
    }
#pragma unroll
    for (int ni = 0; ni < 4; ++ni) {
      int rb = (waveN + ni * 16 + n16) * TS + kbB;
      v8h lo = *(const v8h*)&Bt[cur][rb];
      v8h hi = *(const v8h*)&Bt[cur][rb + 16];
#pragma unroll
      for (int i = 0; i < 8; ++i) { bfr[ni][i] = lo[i]; bfr[ni][8 + i] = hi[i]; }
    }

    // ---- 8 x v_wmma_f32_16x16x32_f16 ----
#pragma unroll
    for (int mi = 0; mi < 2; ++mi)
#pragma unroll
      for (int ni = 0; ni < 4; ++ni)
        acc[mi][ni] = __builtin_amdgcn_wmma_f32_16x16x32_f16(
            false, afr[mi], false, bfr[ni], (short)0, acc[mi][ni], false, false);

    __syncthreads();  // all reads of `cur` done before it is restaged
    cur ^= 1;
  }

  // ---- epilogue: C/D layout = lane n=lane&15, row = v + 8*(lane>>4) ----
  const int mrow0 = (lane >> 4) * 8;
#pragma unroll
  for (int mi = 0; mi < 2; ++mi) {
#pragma unroll
    for (int ni = 0; ni < 4; ++ni) {
      int col = blockN + waveN + ni * 16 + n16;
      float bv = bias ? bias[col] : 0.f;
#pragma unroll
      for (int v = 0; v < 8; ++v) {
        int row   = blockM + waveM + mi * 16 + mrow0 + v;
        float val = acc[mi][ni][v] * alpha + bv;
        if (MODE == 0) {
          ((half_t*)outp)[(size_t)row * ldo + col] = (half_t)val;
        } else if (MODE == 1) {
          int bb = row >> 12, nn = row & (HW - 1);
          ((half_t*)outp)[(size_t)bb * CC * HW + (size_t)col * HW + nn] = (half_t)val;
        } else {
          int bb = row >> 12, nn = row & (HW - 1);
          size_t oi = ((size_t)bb * CC + col) * HW + nn;
          ((float*)outp)[oi] = residual[oi] + val;
        }
      }
    }
  }
}

// =====================================================================
// Kernel 5: row softmax over 4096 f16 values (in place), f32 math.
// =====================================================================
__global__ __launch_bounds__(256) void softmax_rows(half_t* __restrict__ S) {
  half_t* p = S + (size_t)blockIdx.x * HW;
  float v[16];
  float mx = -1e30f;
#pragma unroll
  for (int i = 0; i < 16; ++i) {
    v[i] = (float)p[threadIdx.x + i * 256];
    mx = fmaxf(mx, v[i]);
  }
  __shared__ float red[256];
  red[threadIdx.x] = mx; __syncthreads();
  for (int st = 128; st > 0; st >>= 1) {
    if (threadIdx.x < st) red[threadIdx.x] = fmaxf(red[threadIdx.x], red[threadIdx.x + st]);
    __syncthreads();
  }
  mx = red[0]; __syncthreads();
  float s = 0.f;
#pragma unroll
  for (int i = 0; i < 16; ++i) { v[i] = __expf(v[i] - mx); s += v[i]; }
  red[threadIdx.x] = s; __syncthreads();
  for (int st = 128; st > 0; st >>= 1) {
    if (threadIdx.x < st) red[threadIdx.x] += red[threadIdx.x + st];
    __syncthreads();
  }
  float inv = 1.f / red[0];
#pragma unroll
  for (int i = 0; i < 16; ++i) p[threadIdx.x + i * 256] = (half_t)(v[i] * inv);
}

// =====================================================================
// Host orchestration
// =====================================================================
extern "C" void kernel_launch(void* const* d_in, const int* in_sizes, int n_in,
                              void* d_out, int out_size, void* d_ws, size_t ws_size,
                              hipStream_t stream) {
  const float* x     = (const float*)d_in[0];
  const float* gamma = (const float*)d_in[1];
  const float* beta  = (const float*)d_in[2];
  const float* wq = (const float*)d_in[3]; const float* bq = (const float*)d_in[4];
  const float* wk = (const float*)d_in[5]; const float* bk = (const float*)d_in[6];
  const float* wv = (const float*)d_in[7]; const float* bv = (const float*)d_in[8];
  const float* wp = (const float*)d_in[9]; const float* bp = (const float*)d_in[10];

  // ---- workspace carve-up (~74 MB; attention matrix reused per batch) ----
  char* ws = (char*)d_ws;
  size_t off = 0;
  auto carve = [&](size_t bytes) { void* p = ws + off; off += (bytes + 255) & ~(size_t)255; return p; };
  float*  meanrstd = (float*)carve(BN * GRP * 2 * sizeof(float));
  half_t* wq16 = (half_t*)carve((size_t)CC * CC * 2);
  half_t* wk16 = (half_t*)carve((size_t)CC * CC * 2);
  half_t* wv16 = (half_t*)carve((size_t)CC * CC * 2);
  half_t* wp16 = (half_t*)carve((size_t)CC * CC * 2);
  half_t* T  = (half_t*)carve((size_t)BN * HW * CC * 2);   // normalized acts (b,n,c)
  half_t* Q  = (half_t*)carve((size_t)BN * HW * CC * 2);
  half_t* Kb = (half_t*)carve((size_t)BN * HW * CC * 2);
  half_t* Vt = (half_t*)carve((size_t)BN * CC * HW * 2);   // V transposed: (b,c,n)
  half_t* S  = (half_t*)carve((size_t)HW * HW * 2);        // one batch of scores/probs
  half_t* O  = (half_t*)carve((size_t)BN * HW * CC * 2);   // attention output (b,n,c)

  // ---- 1. GroupNorm stats + apply ----
  gn_stats<<<BN * GRP, 256, 0, stream>>>(x, meanrstd);
  {
    size_t ntot = (size_t)BN * HW * CC;
    gn_apply<<<(unsigned)((ntot + 255) / 256), 256, 0, stream>>>(x, gamma, beta, meanrstd, T);
  }

  // ---- 2. weights to f16 ----
  const int wn = CC * CC;
  f32_to_f16<<<(wn + 255) / 256, 256, 0, stream>>>(wq, wq16, wn);
  f32_to_f16<<<(wn + 255) / 256, 256, 0, stream>>>(wk, wk16, wn);
  f32_to_f16<<<(wn + 255) / 256, 256, 0, stream>>>(wv, wv16, wn);
  f32_to_f16<<<(wn + 255) / 256, 256, 0, stream>>>(wp, wp16, wn);

  // ---- 3. QKV projections (M = B*HW = 8192, N = 512, K = 512) ----
  dim3 gQKV((BN * HW) / 128, CC / 128);
  gemm_wmma<0><<<gQKV, 256, 0, stream>>>(T, CC, wq16, CC, bq, 1.f, Q,  CC, CC, nullptr);
  gemm_wmma<0><<<gQKV, 256, 0, stream>>>(T, CC, wk16, CC, bk, 1.f, Kb, CC, CC, nullptr);
  gemm_wmma<1><<<gQKV, 256, 0, stream>>>(T, CC, wv16, CC, bv, 1.f, Vt, 0,  CC, nullptr);

  // ---- 4. attention per batch (S buffer reused; stream serializes) ----
  const float scale = 0.04419417382415922f;  // 512^-0.5
  for (int b = 0; b < BN; ++b) {
    const size_t bo = (size_t)b * HW * CC;
    dim3 gS(HW / 128, HW / 128);  // scores: M=4096, N=4096, K=512
    gemm_wmma<0><<<gS, 256, 0, stream>>>(Q + bo, CC, Kb + bo, CC, nullptr, scale, S, HW, CC, nullptr);
    softmax_rows<<<HW, 256, 0, stream>>>(S);
    dim3 gO(HW / 128, CC / 128);  // O = P @ V: M=4096, N=512, K=4096
    gemm_wmma<0><<<gO, 256, 0, stream>>>(S, HW, Vt + (size_t)b * CC * HW, HW, nullptr, 1.f,
                                         O + bo, CC, HW, nullptr);
  }

  // ---- 5. output projection + residual into NCHW f32 d_out ----
  dim3 gP((BN * HW) / 128, CC / 128);
  gemm_wmma<2><<<gP, 256, 0, stream>>>(O, CC, wp16, CC, bp, 1.f, d_out, 0, CC, x);
}
